// SparseNet_12489764897163
// MI455X (gfx1250) — matrix operand order repaired and checked by reading, not compiled, not run on hardware
//
#include <hip/hip_runtime.h>
#include <math.h>

// ---------------------------------------------------------------------------
// Fully-fused SparseNet forward for MI455X (gfx1250), bf16 WMMA for conv1.
// One block = one image. 256 threads = 8 waves (wave32).
// Deep-pipelined HBM load stage (16 outstanding b128 loads per thread).
// ---------------------------------------------------------------------------

#define NEGF (-1e30f)

typedef __bf16 bf16_t;
typedef bf16_t v16bf __attribute__((ext_vector_type(16)));
typedef float  v8f   __attribute__((ext_vector_type(8)));

struct Frag32 { unsigned w[8]; };                  // 8 dwords = 16 bf16
struct alignas(16) U128 { unsigned x[4]; };
struct alignas(8)  U64  { unsigned x[2]; };

__device__ __forceinline__ unsigned f2bf(float f) { // f32 -> bf16 (RNE)
    unsigned u = __float_as_uint(f);
    return (u + 0x7FFFu + ((u >> 16) & 1u)) >> 16;
}

// ---- LDS layout (bytes) ----------------------------------------------------
// image region is reused for W2 after conv1 is done.
// h1 and ldsM are padded to 448 rows so WMMA C-tiles store unconditionally.
#define OFF_X    0           // bf16 image [22][22][200] = 193600 B (alias: W2 f32, 51200 B)
#define OFF_B    193600      // W1 fragments: 25 chunks * 8 regs * 32 lanes * 4B = 25600 B
#define OFF_M    219200      // mask  f32 [448]       = 1792 B   (441 real + 7 pad zeros)
#define OFF_H1   220992      // h1    f32 [448][16]   = 28672 B  (441 real + 7 pad rows)
#define OFF_P1   249664      // p1    f32 [49][16]    = 3136 B
#define OFF_M1   252800      // m1    f32 [49]        = 196 B
#define OFF_H2   253008      // h2    f32 [9][32]     = 1152 B
#define OFF_M2   254160      // m2    f32 [9]         = 36 B
#define OFF_PL   254208      // pooled f32 [32]       = 128 B
#define SMEM_BYTES 254336

extern "C" __global__ void __launch_bounds__(256, 1)
sparse_net_fused(const float* __restrict__ xg, const int* __restrict__ maskg,
                 const float* __restrict__ w1g, const float* __restrict__ w2g,
                 const float* __restrict__ wling, const float* __restrict__ bling,
                 float* __restrict__ outg)
{
    extern __shared__ char smem[];
    const int b    = blockIdx.x;
    const int tid  = threadIdx.x;
    const int lane = tid & 31;
    // readfirstlane makes the wave id an SGPR -> tile guards become scalar
    // branches instead of exec-masked vector control flow.
    const int wv   = __builtin_amdgcn_readfirstlane(tid >> 5);   // 0..7
    const int half = lane >> 4;       // 0: lanes 0-15, 1: lanes 16-31

    float*    ldsM = (float*)(smem + OFF_M);
    unsigned* ldsB = (unsigned*)(smem + OFF_B);
    float*    h1   = (float*)(smem + OFF_H1);
    float*    p1   = (float*)(smem + OFF_P1);
    float*    m1   = (float*)(smem + OFF_M1);
    float*    h2   = (float*)(smem + OFF_H2);
    float*    m2   = (float*)(smem + OFF_M2);
    float*    pl   = (float*)(smem + OFF_PL);

    // ---- Stage 0: mask -> LDS (448, pad=0), zero image pads, pack W1 -------
    for (int s = tid; s < 448; s += 256)
        ldsM[s] = (s < 441) ? (float)maskg[b * 441 + s] : 0.0f;

    // pad row i==21 : 22*400 B = 550 x 16B ; pad col j==21 rows 0..20 : 21*25 x 16B
    for (int p = tid; p < 550 + 21 * 25; p += 256) {
        U128* dst;
        if (p < 550) {
            dst = (U128*)(smem + (21 * 22) * 400 + p * 16);
        } else {
            int q = p - 550; int i = q / 25; int k = q - i * 25;
            dst = (U128*)(smem + (i * 22 + 21) * 400 + k * 16);
        }
        *dst = U128{{0u, 0u, 0u, 0u}};
    }

    // W1 [800][16] f32 -> bf16 in B-fragment order:
    // dword d = c*256 + j*32 + l holds B[k][n],B[k+1][n], k = 32c + 16*(l>>4) + 2j, n = l&15
    for (int d = tid; d < 6400; d += 256) {
        int l = d & 31; int j = (d >> 5) & 7; int c = d >> 8;
        int n = l & 15;
        int k = c * 32 + ((l >> 4) << 4) + 2 * j;
        unsigned lo = f2bf(w1g[k * 16 + n]);
        unsigned hi = f2bf(w1g[(k + 1) * 16 + n]);
        ldsB[d] = lo | (hi << 16);
    }
    __syncthreads();

    // ---- Stage 1: x * mask -> bf16 image [22][22][200] (one HBM pass) ------
    // 22050 float4 per image. 5 full groups of 256*16, then remainder.
    // 16 independent global b128 loads are issued before any consume so each
    // wave keeps 16 loads outstanding (HBM latency hiding).
    {
        const float4* xg4 = (const float4*)xg + (size_t)b * 22050;
        int base = tid;
        #pragma unroll 1
        for (int g = 0; g < 5; ++g) {
            float4 buf[16];
            #pragma unroll
            for (int u = 0; u < 16; ++u) buf[u] = xg4[base + u * 256];
            #pragma unroll
            for (int u = 0; u < 16; ++u) {
                int f = base + u * 256;
                int pix = f / 50; int c4 = f - pix * 50;
                float mv = ldsM[pix];
                unsigned d0 = f2bf(buf[u].x * mv) | (f2bf(buf[u].y * mv) << 16);
                unsigned d1 = f2bf(buf[u].z * mv) | (f2bf(buf[u].w * mv) << 16);
                int i = pix / 21; int j = pix - i * 21;
                *(U64*)(smem + (i * 22 + j) * 400 + c4 * 8) = U64{{d0, d1}};
            }
            base += 4096;
        }
        for (int f = 5 * 4096 + tid; f < 22050; f += 256) {
            int pix = f / 50; int c4 = f - pix * 50;
            float4 v = xg4[f];
            float mv = ldsM[pix];
            unsigned d0 = f2bf(v.x * mv) | (f2bf(v.y * mv) << 16);
            unsigned d1 = f2bf(v.z * mv) | (f2bf(v.w * mv) << 16);
            int i = pix / 21; int j = pix - i * 21;
            *(U64*)(smem + (i * 22 + j) * 400 + c4 * 8) = U64{{d0, d1}};
        }
    }
    __syncthreads();

    // ---- Stage 2: conv1 as implicit GEMM, v_wmma_f32_16x16x32_bf16 ---------
    // A: [441(+pad) x 800] im2col of masked image, B: [800 x 16] = W1.
    // 32 M-tiles (28 real + 4 pad), 4 tiles per wave.
    v8f acc[4];
    #pragma unroll
    for (int t = 0; t < 4; ++t) {
        #pragma unroll
        for (int r = 0; r < 8; ++r) acc[t][r] = 0.0f;
    }

    int rb[4];                 // per-lane row base byte offset into image
    #pragma unroll
    for (int t = 0; t < 4; ++t) {
        int row  = (wv * 4 + t) * 16 + (lane & 15);
        int rowc = row > 440 ? 440 : row;           // clamp pad rows
        int i = rowc / 21, j = rowc - i * 21;
        rb[t] = (i * 22 + j) * 400;
    }

    for (int c = 0; c < 25; ++c) {                  // K chunks of 32 (K = 800)
        int k1 = c * 32 + (half << 4);              // this lane's first K
        int k2 = k1 + 8;
        int blk1 = k1 / 200; int ci1 = k1 - blk1 * 200;   // (di,dj) block, channel
        int blk2 = k2 / 200; int ci2 = k2 - blk2 * 200;
        int off1 = ((blk1 >> 1) * 22 + (blk1 & 1)) * 400 + ci1 * 2;
        int off2 = ((blk2 >> 1) * 22 + (blk2 & 1)) * 400 + ci2 * 2;

        Frag32 bf;                                   // B fragment (shared by all tiles)
        #pragma unroll
        for (int j = 0; j < 8; ++j)
            bf.w[j] = ldsB[c * 256 + j * 32 + lane];
        v16bf bvec = __builtin_bit_cast(v16bf, bf);

        #pragma unroll
        for (int t = 0; t < 4; ++t) {
            U128 lo = *(const U128*)(smem + rb[t] + off1);   // ds_load_b128
            U128 hi = *(const U128*)(smem + rb[t] + off2);   // ds_load_b128
            Frag32 af;
            #pragma unroll
            for (int q = 0; q < 4; ++q) { af.w[q] = lo.x[q]; af.w[4 + q] = hi.x[q]; }
            v16bf avec = __builtin_bit_cast(v16bf, af);
            acc[t] = __builtin_amdgcn_wmma_f32_16x16x32_bf16(
                         false, avec, false, bvec, (short)0, acc[t], false, false);
        }
    }

    // scatter C frags -> h1 (masked).  C layout: reg r, lanes 0-15: M=r, 16-31: M=8+r
    // tb is an SGPR (wv uniform); rows 441..447 land in h1/ldsM pad -> no
    // per-row predication.  This lane's 8 rows are consecutive, so the 8 mask
    // values are prefetched as two ds_load_b128 (one wait instead of eight).
    #pragma unroll
    for (int t = 0; t < 4; ++t) {
        int tb  = (wv * 4 + t) * 16;
        int col = lane & 15;
        if (tb < 441) {
            const float* mrow = &ldsM[tb + (half << 3)];     // 16B-aligned
            U128 ma = *(const U128*)(mrow);
            U128 mb = *(const U128*)(mrow + 4);
            float mv[8];
            #pragma unroll
            for (int q = 0; q < 4; ++q) {
                mv[q]     = __uint_as_float(ma.x[q]);
                mv[4 + q] = __uint_as_float(mb.x[q]);
            }
            #pragma unroll
            for (int r = 0; r < 8; ++r) {
                int row = tb + r + (half << 3);
                h1[row * 16 + col] = acc[t][r] * mv[r];
            }
        }
    }
    __syncthreads();

    // ---- Stage 3: masked 3x3/3 maxpool (21->7) + stage W2 into LDS ---------
    {   // W2 [5][5][16][32] f32 = 12800 floats into image-alias region
        const float4* w2g4 = (const float4*)w2g;
        float4* w2l4 = (float4*)(smem + OFF_X);
        for (int f = tid; f < 3200; f += 256) w2l4[f] = w2g4[f];
    }
    for (int idx = tid; idx < 784; idx += 256) {     // 49 sites * 16 ch
        int ch = idx & 15; int s = idx >> 4;
        int pi = s / 7, pj = s - pi * 7;
        float best = NEGF, mo = 0.0f;
        #pragma unroll
        for (int wi = 0; wi < 3; ++wi)
            #pragma unroll
            for (int wj = 0; wj < 3; ++wj) {
                int ii = pi * 3 + wi, jj = pj * 3 + wj;
                float mm = ldsM[ii * 21 + jj];
                mo = fmaxf(mo, mm);
                float hv = h1[(ii * 21 + jj) * 16 + ch];
                best = fmaxf(best, mm > 0.0f ? hv : NEGF);
            }
        p1[s * 16 + ch] = mo > 0.0f ? best : 0.0f;
        if (ch == 0) m1[s] = mo;
    }
    __syncthreads();

    // ---- Stage 4: conv2 5x5 VALID (7->3), 16->32, masked --------------------
    const float* w2l = (const float*)(smem + OFF_X);
    for (int idx = tid; idx < 288; idx += 256) {     // 9 sites * 32 ch
        int co = idx & 31; int s = idx >> 5;
        int oi = s / 3, oj = s - oi * 3;
        float sum = 0.0f, mo = 0.0f;
        #pragma unroll
        for (int ki = 0; ki < 5; ++ki)
            for (int kj = 0; kj < 5; ++kj) {
                int ps = (oi + ki) * 7 + (oj + kj);
                mo = fmaxf(mo, m1[ps]);
                const float* prow = &p1[ps * 16];
                const float* wrow = &w2l[((ki * 5 + kj) * 16) * 32 + co];
                #pragma unroll
                for (int ci = 0; ci < 16; ++ci)
                    sum += prow[ci] * wrow[ci * 32];
            }
        h2[s * 32 + co] = sum * mo;                  // mo in {0,1}
        if (co == 0) m2[s] = mo;
    }
    __syncthreads();

    // ---- Stage 5: masked 3x3 maxpool (3->1) --------------------------------
    if (tid < 32) {
        float best = NEGF, mo = 0.0f;
        #pragma unroll
        for (int s = 0; s < 9; ++s) {
            float mm = m2[s];
            mo = fmaxf(mo, mm);
            best = fmaxf(best, mm > 0.0f ? h2[s * 32 + tid] : NEGF);
        }
        pl[tid] = mo > 0.0f ? best : 0.0f;
    }
    __syncthreads();

    // ---- Stage 6: Linear(32,2) + softmax -----------------------------------
    if (tid == 0) {
        float y0 = bling[0], y1 = bling[1];
        #pragma unroll
        for (int c = 0; c < 32; ++c) {
            float v = pl[c];
            y0 += v * wling[c * 2 + 0];
            y1 += v * wling[c * 2 + 1];
        }
        float mx = fmaxf(y0, y1);
        float e0 = expf(y0 - mx), e1 = expf(y1 - mx);
        float inv = 1.0f / (e0 + e1);
        outg[b * 2 + 0] = e0 * inv;
        outg[b * 2 + 1] = e1 * inv;
    }
}

extern "C" void kernel_launch(void* const* d_in, const int* in_sizes, int n_in,
                              void* d_out, int out_size, void* d_ws, size_t ws_size,
                              hipStream_t stream) {
    (void)n_in; (void)d_ws; (void)ws_size; (void)out_size;
    const int B = in_sizes[0] / (21 * 21 * 200);     // 1024
    hipFuncSetAttribute(reinterpret_cast<const void*>(sparse_net_fused),
                        hipFuncAttributeMaxDynamicSharedMemorySize, SMEM_BYTES);
    sparse_net_fused<<<dim3(B), dim3(256), SMEM_BYTES, stream>>>(
        (const float*)d_in[0], (const int*)d_in[1], (const float*)d_in[2],
        (const float*)d_in[3], (const float*)d_in[4], (const float*)d_in[5],
        (float*)d_out);
}
